// Decoder_14568529068506
// MI455X (gfx1250) — compile-verified, matrix operand order checked by reading, not run on hardware
//
#include <hip/hip_runtime.h>

typedef __attribute__((ext_vector_type(16))) _Float16 v16h;
typedef __attribute__((ext_vector_type(8)))  _Float16 v8h;
typedef __attribute__((ext_vector_type(8)))  float    v8f;

#define S_   32
#define P_   48
#define T_   8
#define E_   64
#define H_   64
#define D1_  512
#define D2_  1024
#define K1_  576                 // T*E + H : layer-1 input width
#define XW_  584                 // padded LDS row stride (halfs) for X
#define YW_  520                 // padded LDS row stride (halfs) for Y
#define NTHREADS_ 256
#define NWAVES_   8
#define BTOT_ (S_ * P_)

#define PPB_  2                  // pedestrians (i) per workgroup
#define ROWS_ (PPB_ * P_)        // 96 rows per workgroup
#define MT_   (ROWS_ / 16)       // 6 M-tiles

// ---------------------------------------------------------------------------
// WMMA fragment loaders for V_WMMA_F32_16X16X32_F16 (wave32).
// 16-bit A (16x32) per-lane layout (ISA 7.12.2): lanes 0-15 hold K=0..7 in
// VGPR0-3 and K=16..23 in VGPR4-7; lanes 16-31 hold K=8..15 and K=24..31.
// Each lane needs two contiguous 8-half (16B) runs at offsets {base, 16+base},
// base = (lane<16) ? 0 : 8. B mirrors A with N = lane%16.
// ---------------------------------------------------------------------------

__device__ __forceinline__ v16h load_a_frag_lds(const _Float16* row, int kk, int base) {
  const _Float16* p = row + kk * 32;
  v8h lo = *(const v8h*)(p + base);        // 16B aligned ds_load_b128
  v8h hi = *(const v8h*)(p + 16 + base);
  v16h a;
#pragma unroll
  for (int q = 0; q < 8; ++q) { a[q] = lo[q]; a[8 + q] = hi[q]; }
  return a;
}

template <bool HALFW>
__device__ __forceinline__ v16h load_b_frag(const void* wbase, size_t rowoff,
                                            int kk, int base) {
  v16h b;
  if constexpr (HALFW) {
    const _Float16* p = (const _Float16*)wbase + rowoff + kk * 32;
    v8h lo = *(const v8h*)(p + base);      // global_load_b128, no cvt
    v8h hi = *(const v8h*)(p + 16 + base);
#pragma unroll
    for (int q = 0; q < 8; ++q) { b[q] = lo[q]; b[8 + q] = hi[q]; }
  } else {
    const float* p = (const float*)wbase + rowoff + kk * 32;
#pragma unroll
    for (int q = 0; q < 8; ++q) b[q] = (_Float16)p[base + q];
#pragma unroll
    for (int q = 0; q < 8; ++q) b[8 + q] = (_Float16)p[16 + base + q];
  }
  return b;
}

// ---------------------------------------------------------------------------
// Pre-pass: convert W1 (512x576) and W2 (1024x512) to f16 in workspace.
// ---------------------------------------------------------------------------
__global__ __launch_bounds__(256)
void cvt_weights_kernel(const float* __restrict__ W1, const float* __restrict__ W2,
                        _Float16* __restrict__ out) {
  const int n1 = D1_ * K1_;
  const int n2 = D2_ * D1_;
  int i = blockIdx.x * 256 + threadIdx.x;
  if (i < n1)            out[i] = (_Float16)W1[i];
  else if (i < n1 + n2)  out[i] = (_Float16)W2[i - n1];
}

// ---------------------------------------------------------------------------
// Ping-pong pipelined GEMM tile loop (NK even): buffers {a0,b0} / {a1,b1}
// alternate, so fragments for kk+1 are in flight while the six WMMAs of kk
// issue — and no cross-iteration register copies are needed.
// ---------------------------------------------------------------------------
template <bool HALFW, int NK>
__device__ __forceinline__ void gemm_tile(const _Float16* __restrict__ Asrc,
                                          int astride, const void* __restrict__ Wv,
                                          size_t browoff, int ln16, int base,
                                          v8f acc[MT_]) {
  static_assert((NK & 1) == 0, "NK must be even for ping-pong pipelining");
  v16h a0[MT_], a1[MT_];
  v16h b0, b1;

  b0 = load_b_frag<HALFW>(Wv, browoff, 0, base);
#pragma unroll
  for (int mt = 0; mt < MT_; ++mt)
    a0[mt] = load_a_frag_lds(Asrc + (size_t)(mt * 16 + ln16) * astride, 0, base);

  for (int kk = 0; kk < NK; kk += 2) {
    // stage kk+1 while computing kk
    b1 = load_b_frag<HALFW>(Wv, browoff, kk + 1, base);
#pragma unroll
    for (int mt = 0; mt < MT_; ++mt)
      a1[mt] =
          load_a_frag_lds(Asrc + (size_t)(mt * 16 + ln16) * astride, kk + 1, base);
#pragma unroll
    for (int mt = 0; mt < MT_; ++mt)
      acc[mt] = __builtin_amdgcn_wmma_f32_16x16x32_f16(
          false, a0[mt], false, b0, (short)0, acc[mt], false, false);

    // stage kk+2 while computing kk+1
    if (kk + 2 < NK) {
      b0 = load_b_frag<HALFW>(Wv, browoff, kk + 2, base);
#pragma unroll
      for (int mt = 0; mt < MT_; ++mt)
        a0[mt] =
            load_a_frag_lds(Asrc + (size_t)(mt * 16 + ln16) * astride, kk + 2, base);
    }
#pragma unroll
    for (int mt = 0; mt < MT_; ++mt)
      acc[mt] = __builtin_amdgcn_wmma_f32_16x16x32_f16(
          false, a1[mt], false, b1, (short)0, acc[mt], false, false);
  }
}

// ---------------------------------------------------------------------------
// One workgroup per (scene s, pedestrian pair {2ip, 2ip+1}).
// 96 j-rows -> fused GEMM -> GEMM -> per-ped max_j.
// ---------------------------------------------------------------------------
template <bool HALFW>
__global__ __launch_bounds__(NTHREADS_)
void pool_net_kernel(const float* __restrict__ h_states,   // (B, H)
                     const float* __restrict__ traj,       // (20, B, 2)
                     const float* __restrict__ tw,         // (S, P*P, 2, T)
                     const float* __restrict__ W_se,       // (512, 16)
                     const float* __restrict__ b_se,       // (512)
                     const void*  __restrict__ W1v,        // (512, 576) f32|f16
                     const float* __restrict__ b1,         // (512)
                     const void*  __restrict__ W2v,        // (1024, 512) f32|f16
                     const float* __restrict__ b2,         // (1024)
                     float* __restrict__ out)              // (B, 1024)
{
  extern __shared__ char smem[];
  _Float16* Xs   = (_Float16*)smem;                                  // ROWS_*XW_
  _Float16* Ys   = (_Float16*)(smem + ROWS_ * XW_ * 2);              // ROWS_*YW_
  float*    rels = (float*)(smem + ROWS_ * XW_ * 2 + ROWS_ * YW_ * 2); // ROWS_*16

  const int blk  = blockIdx.x;             // 0..767
  const int s    = blk / (P_ / PPB_);
  const int ip   = blk - s * (P_ / PPB_);
  const int i0   = ip * PPB_;              // first ped of the pair
  const int tid  = threadIdx.x;
  const int lane = tid & 31;
  const int wave = tid >> 5;
  const int ln16 = lane & 15;
  const int base = (lane & 16) ? 8 : 0;

  // ---- phase 0a: rel[ii][j][k] = obs[s,j] - obs[s,i0+ii], k = t*2+d ----------
  for (int idx = tid; idx < ROWS_ * 16; idx += NTHREADS_) {
    const int r  = idx >> 4;               // ii*48 + j
    const int k  = idx & 15;
    const int ii = r / P_;
    const int j  = r - ii * P_;
    const int t  = k >> 1, d = k & 1;
    const size_t bo = (size_t)t * BTOT_ * 2;
    rels[idx] = traj[bo + (size_t)(s * P_ + j) * 2 + d]
              - traj[bo + (size_t)(s * P_ + i0 + ii) * 2 + d];
  }
  __syncthreads();

  // ---- phase 0b: X[r][c] = { tw * (rel @ W_se^T + b_se) | h_j } as f16 -------
  for (int idx = tid; idx < ROWS_ * K1_; idx += NTHREADS_) {
    const int r  = idx / K1_;
    const int c  = idx - r * K1_;
    const int ii = r / P_;
    const int j  = r - ii * P_;
    float v;
    if (c < T_ * E_) {
      const int t = c >> 6, e = c & 63;
      const float* wr = W_se + (size_t)c * 16;
      const float* rr = rels + r * 16;
      float acc = b_se[c];
#pragma unroll
      for (int k = 0; k < 16; ++k) acc = fmaf(rr[k], wr[k], acc);
      const float wv =
          tw[(((size_t)s * (P_ * P_) + (size_t)(i0 + ii) * P_ + j) * 2 + (e & 1)) * T_ + t];
      v = wv * acc;
    } else {
      v = h_states[(size_t)(s * P_ + j) * H_ + (c - T_ * E_)];
    }
    Xs[(size_t)r * XW_ + c] = (_Float16)v;
  }
  __syncthreads();

  // ---- layer 1: Y = relu(X @ W1^T + b1)   [96x576]x[576x512] -----------------
  for (int nt = wave; nt < D1_ / 16; nt += NWAVES_) {
    const size_t browoff = (size_t)(nt * 16 + ln16) * K1_;
    // prefetch next owned N-tile's weight row into cache while we compute
    if (nt + NWAVES_ < D1_ / 16) {
      const size_t nxt = (size_t)((nt + NWAVES_) * 16 + ln16) * K1_;
      if constexpr (HALFW) __builtin_prefetch((const _Float16*)W1v + nxt, 0, 3);
      else                 __builtin_prefetch((const float*)W1v + nxt, 0, 3);
    }
    const float bias = b1[nt * 16 + ln16];
    v8f acc[MT_];
#pragma unroll
    for (int mt = 0; mt < MT_; ++mt)
#pragma unroll
      for (int r = 0; r < 8; ++r) acc[mt][r] = bias;       // bias folded into C

    gemm_tile<HALFW, K1_ / 32>(Xs, XW_, W1v, browoff, ln16, base, acc);

    // relu -> f16 -> LDS. Lane holds N=ln16; VGPR r is row base+r of the tile.
#pragma unroll
    for (int mt = 0; mt < MT_; ++mt)
#pragma unroll
      for (int r = 0; r < 8; ++r) {
        const float v = fmaxf(acc[mt][r], 0.f);
        Ys[(size_t)(mt * 16 + base + r) * YW_ + nt * 16 + ln16] = (_Float16)v;
      }
  }
  __syncthreads();

  // ---- layer 2 + max_j: out = max_rows relu(Y @ W2^T + b2), per ped ----------
  for (int nt = wave; nt < D2_ / 16; nt += NWAVES_) {
    const size_t browoff = (size_t)(nt * 16 + ln16) * D1_;
    if (nt + NWAVES_ < D2_ / 16) {
      const size_t nxt = (size_t)((nt + NWAVES_) * 16 + ln16) * D1_;
      if constexpr (HALFW) __builtin_prefetch((const _Float16*)W2v + nxt, 0, 3);
      else                 __builtin_prefetch((const float*)W2v + nxt, 0, 3);
    }
    const float bias = b2[nt * 16 + ln16];
    v8f acc[MT_];
#pragma unroll
    for (int mt = 0; mt < MT_; ++mt)
#pragma unroll
      for (int r = 0; r < 8; ++r) acc[mt][r] = bias;

    gemm_tile<HALFW, D1_ / 32>(Ys, YW_, W2v, browoff, ln16, base, acc);

    // fused relu + row-max, separately for each pedestrian's 3 M-tiles
    float cmax[PPB_];
#pragma unroll
    for (int ii = 0; ii < PPB_; ++ii) {
      float m = -3.402823466e38f;
#pragma unroll
      for (int mt = ii * (MT_ / PPB_); mt < (ii + 1) * (MT_ / PPB_); ++mt)
#pragma unroll
        for (int r = 0; r < 8; ++r)
          m = fmaxf(m, fmaxf(acc[mt][r], 0.f));
      m = fmaxf(m, __shfl_xor(m, 16, 32));   // rows 0-7 vs 8-15 of each tile
      cmax[ii] = m;
    }
    if (lane < 16) {
#pragma unroll
      for (int ii = 0; ii < PPB_; ++ii)
        out[(size_t)(s * P_ + i0 + ii) * D2_ + nt * 16 + ln16] = cmax[ii];
    }
  }
}

// ---------------------------------------------------------------------------
extern "C" void kernel_launch(void* const* d_in, const int* in_sizes, int n_in,
                              void* d_out, int out_size, void* d_ws, size_t ws_size,
                              hipStream_t stream) {
  const float* h_states = (const float*)d_in[0];
  // d_in[1] seq_start_end (uniform scenes -> unused), d_in[2] end_pos (unused)
  const float* traj = (const float*)d_in[3];
  const float* tw   = (const float*)d_in[4];
  // d_in[5] mlp_pre_pool_dim_0 scalar (== 512, baked in)
  const float* W_se = (const float*)d_in[6];
  const float* b_se = (const float*)d_in[7];
  const float* W1   = (const float*)d_in[8];
  const float* b1   = (const float*)d_in[9];
  const float* W2   = (const float*)d_in[10];
  const float* b2   = (const float*)d_in[11];
  float* out = (float*)d_out;

  const size_t shmem = (size_t)ROWS_ * XW_ * 2   // X   (f16)  112128 B
                     + (size_t)ROWS_ * YW_ * 2   // Y   (f16)   99840 B
                     + (size_t)ROWS_ * 16 * 4;   // rel (f32)    6144 B

  dim3 grid(BTOT_ / PPB_), block(NTHREADS_);

  const size_t n1 = (size_t)D1_ * K1_;
  const size_t n2 = (size_t)D2_ * D1_;
  const size_t need = (n1 + n2) * sizeof(_Float16);

  if (ws_size >= need) {
    // pre-convert weights to f16 once, then run the WMMA kernel on f16 weights
    _Float16* wh = (_Float16*)d_ws;
    const int total = (int)(n1 + n2);
    hipLaunchKernelGGL(cvt_weights_kernel, dim3((total + 255) / 256), dim3(256),
                       0, stream, W1, W2, wh);
    hipLaunchKernelGGL((pool_net_kernel<true>), grid, block, shmem, stream,
                       h_states, traj, tw, W_se, b_se,
                       (const void*)wh, b1, (const void*)(wh + n1), b2, out);
  } else {
    // fallback: convert f32 weights to f16 on the fly inside the hot loop
    hipLaunchKernelGGL((pool_net_kernel<false>), grid, block, shmem, stream,
                       h_states, traj, tw, W_se, b_se,
                       (const void*)W1, b1, (const void*)W2, b2, out);
  }
}
